// SeqLinear_69844758167722
// MI455X (gfx1250) — compile-verified, matrix operand order checked by reading, not run on hardware
//
#include <hip/hip_runtime.h>
#include <hip/hip_bf16.h>

typedef __attribute__((ext_vector_type(16))) __bf16 v16bf;
typedef __attribute__((ext_vector_type(8)))  float  v8f;

#define D_MODEL   1024
#define N_HEADS   16
#define D_CONVK   4
#define D_STATE   1024
#define D_INNER   1024
#define CONV_DIM  3072
#define D_IN_PROJ 3088
#define BB        4
#define TT        1024
#define BT        (BB*TT)

union Frag { v16bf v; uint4 q[2]; };

// ---------------- fp32 -> bf16 conversion (vectorized, n % 4 == 0) ----------------
__global__ void cvt_bf16_k(const float4* __restrict__ in, uint2* __restrict__ out,
                           int n4) {
    int i = blockIdx.x * blockDim.x + threadIdx.x;
    if (i >= n4) return;
    float4 f = in[i];
    union { __hip_bfloat16 h[4]; uint2 u; } r;
    r.h[0] = __float2bfloat16(f.x);
    r.h[1] = __float2bfloat16(f.y);
    r.h[2] = __float2bfloat16(f.z);
    r.h[3] = __float2bfloat16(f.w);
    out[i] = r.u;
}

// ---------------- WMMA bf16 GEMM:  C[M,N] = A[M,K] * B[N,K]^T ----------------
// A row-major (lda), B row-major (ldb): K contiguous for both.
// One wave computes a 16 x (16*NT) strip; NT is a compile-time constant so the
// hot loop is branch-free. Software double-buffering overlaps the next K-step's
// global_load_b128s with the current K-step's WMMAs.
template <int NT>
__global__ __launch_bounds__(128)
void gemm_bf16_wmma(const __hip_bfloat16* __restrict__ A,
                    const __hip_bfloat16* __restrict__ Bm,
                    float* __restrict__ C,
                    int ntile_base, int K, int lda, int ldb, int ldc) {
    const int lane  = threadIdx.x & 31;
    const int wave  = threadIdx.x >> 5;
    const int mtile = blockIdx.y * 4 + wave;
    const int nt0   = ntile_base + blockIdx.x * NT;

    const int col  = lane & 15;     // row-within-tile (A) / col-within-tile (B,C)
    const int half = lane >> 4;     // K-octet selector

    const __hip_bfloat16* Arow = A + (size_t)(mtile * 16 + col) * lda + half * 8;
    const __hip_bfloat16* Brow[NT];
    #pragma unroll
    for (int j = 0; j < NT; ++j)
        Brow[j] = Bm + (size_t)((nt0 + j) * 16 + col) * ldb + half * 8;

    v8f acc[NT];
    #pragma unroll
    for (int j = 0; j < NT; ++j) acc[j] = (v8f){};

    Frag aCur, bCur[NT];
    aCur.q[0] = *(const uint4*)(Arow);
    aCur.q[1] = *(const uint4*)(Arow + 16);
    #pragma unroll
    for (int j = 0; j < NT; ++j) {
        bCur[j].q[0] = *(const uint4*)(Brow[j]);
        bCur[j].q[1] = *(const uint4*)(Brow[j] + 16);
    }

    const int steps = K >> 5;
    for (int s = 0; s < steps - 1; ++s) {
        const int kn = (s + 1) * 32;
        Frag aNxt, bNxt[NT];
        // issue next K-step loads first ...
        aNxt.q[0] = *(const uint4*)(Arow + kn);
        aNxt.q[1] = *(const uint4*)(Arow + kn + 16);
        #pragma unroll
        for (int j = 0; j < NT; ++j) {
            bNxt[j].q[0] = *(const uint4*)(Brow[j] + kn);
            bNxt[j].q[1] = *(const uint4*)(Brow[j] + kn + 16);
        }
        // ... then compute on current fragments (loads above remain in flight)
        #pragma unroll
        for (int j = 0; j < NT; ++j)
            acc[j] = __builtin_amdgcn_wmma_f32_16x16x32_bf16(
                false, aCur.v, false, bCur[j].v, (short)0, acc[j], false, false);
        aCur = aNxt;
        #pragma unroll
        for (int j = 0; j < NT; ++j) bCur[j] = bNxt[j];
    }
    #pragma unroll
    for (int j = 0; j < NT; ++j)
        acc[j] = __builtin_amdgcn_wmma_f32_16x16x32_bf16(
            false, aCur.v, false, bCur[j].v, (short)0, acc[j], false, false);

    // C/D layout: lane half selects M block (0..7 / 8..15), VGPR r = row, lane&15 = col
    #pragma unroll
    for (int r = 0; r < 8; ++r) {
        const int m = half * 8 + r;
        float* Crow = C + (size_t)(mtile * 16 + m) * ldc;
        #pragma unroll
        for (int j = 0; j < NT; ++j)
            Crow[(nt0 + j) * 16 + col] = acc[j][r];
    }
}

// ---------------- depthwise causal conv (K=4) + gate g = sum(xs*Ii) ----------------
// one block per (b,t); thread c handles channels c, c+1024, c+2048
__global__ __launch_bounds__(1024)
void conv_gate_k(const float* __restrict__ xIOw,
                 const float* __restrict__ conv_w,
                 const float* __restrict__ conv_b,
                 const float* __restrict__ w_base,
                 float* __restrict__ g,
                 float* __restrict__ wh,
                 float* __restrict__ Oo) {
    __shared__ float red[1024];
    const int bt = blockIdx.x;
    const int t  = bt & (TT - 1);
    const int c  = threadIdx.x;

    float as = conv_b[c];
    float ai = conv_b[c + D_STATE];
    float ao = conv_b[c + 2 * D_STATE];
    #pragma unroll
    for (int k = 0; k < D_CONVK; ++k) {
        const int tt = t - (D_CONVK - 1) + k;
        if (tt >= 0) {
            const float* row = xIOw + (size_t)(bt - (D_CONVK - 1) + k) * D_IN_PROJ;
            as += conv_w[(size_t)c * 4 + k]                  * row[c];
            ai += conv_w[(size_t)(c + D_STATE) * 4 + k]      * row[c + D_STATE];
            ao += conv_w[(size_t)(c + 2 * D_STATE) * 4 + k]  * row[c + 2 * D_STATE];
        }
    }
    Oo[(size_t)bt * D_INNER + c] = ao;
    red[c] = as * ai;
    __syncthreads();
    if (c < N_HEADS) {
        float s = 0.f;
        #pragma unroll 8
        for (int d = 0; d < 64; ++d) s += red[c * 64 + d];
        g[(size_t)bt * N_HEADS + c]  = s;
        wh[(size_t)bt * N_HEADS + c] =
            xIOw[(size_t)bt * D_IN_PROJ + CONV_DIM + c] * w_base[c];
    }
}

// ---------------- online-softmax prefix scan ----------------
// a_l = sum_{s<=l} e^{-cs_s} g_s / sum_{s<=l} e^{-cs_s}
// one wave32 per (b,h); wave-level scan of associative state (m, N, D)
__global__ __launch_bounds__(32)
void scan_k(const float* __restrict__ wh, const float* __restrict__ g,
            float* __restrict__ a) {
    const int b = blockIdx.x >> 4;
    const int h = blockIdx.x & 15;
    const int lane = threadIdx.x;

    float cs0 = 0.f, mC = -3.0e38f, nC = 0.f, dC = 0.f;
    for (int chunk = 0; chunk < TT / 32; ++chunk) {
        const int t = chunk * 32 + lane;
        const size_t idx = (size_t)(b * TT + t) * N_HEADS + h;
        const float wv = wh[idx];
        const float gv = g[idx];

        // inclusive prefix sum of wh -> cs within chunk
        float p = wv;
        #pragma unroll
        for (int off = 1; off < 32; off <<= 1) {
            float o = __shfl_up(p, (unsigned)off, 32);
            if (lane >= off) p += o;
        }
        const float cs = cs0 + p;

        // inclusive scan of online-softmax state
        float m = -cs, Nv = gv, Dv = 1.f;
        #pragma unroll
        for (int off = 1; off < 32; off <<= 1) {
            float mo = __shfl_up(m,  (unsigned)off, 32);
            float no = __shfl_up(Nv, (unsigned)off, 32);
            float dd = __shfl_up(Dv, (unsigned)off, 32);
            if (lane >= off) {
                const float M  = fmaxf(m, mo);
                const float e1 = __expf(m - M), e2 = __expf(mo - M);
                Nv = Nv * e1 + no * e2;
                Dv = Dv * e1 + dd * e2;
                m = M;
            }
        }
        // prepend carry from earlier chunks
        const float M2 = fmaxf(m, mC);
        const float e1 = __expf(m - M2), e2 = __expf(mC - M2);
        const float Nn = Nv * e1 + nC * e2;
        const float Dn = Dv * e1 + dC * e2;
        a[idx] = Nn / Dn;

        cs0 = __shfl(cs, 31, 32);
        mC  = __shfl(M2, 31, 32);
        nC  = __shfl(Nn, 31, 32);
        dC  = __shfl(Dn, 31, 32);
    }
}

// ---------------- inner = a * Oo, converted to bf16 for GEMM2 ----------------
__global__ void inner_k(const float* __restrict__ a, const float* __restrict__ Oo,
                        __hip_bfloat16* __restrict__ inner) {
    int i = blockIdx.x * blockDim.x + threadIdx.x;
    if (i >= BT * D_INNER) return;
    const int j  = i & (D_INNER - 1);
    const int bt = i >> 10;
    const int h  = j >> 6;
    inner[i] = __float2bfloat16(a[(size_t)bt * N_HEADS + h] * Oo[i]);
}

extern "C" void kernel_launch(void* const* d_in, const int* in_sizes, int n_in,
                              void* d_out, int out_size, void* d_ws, size_t ws_size,
                              hipStream_t stream) {
    const float* x      = (const float*)d_in[0];
    const float* W_in   = (const float*)d_in[1];
    const float* conv_w = (const float*)d_in[2];
    const float* conv_b = (const float*)d_in[3];
    const float* w_base = (const float*)d_in[4];
    const float* W_out  = (const float*)d_in[5];
    float* out = (float*)d_out;

    char* ws = (char*)d_ws;
    size_t off = 0;
    auto carve = [&](size_t bytes) {
        void* p = ws + off;
        off = (off + bytes + 255) & ~(size_t)255;
        return p;
    };
    __hip_bfloat16* xb     = (__hip_bfloat16*)carve((size_t)BT * D_MODEL * 2);
    __hip_bfloat16* winb   = (__hip_bfloat16*)carve((size_t)D_IN_PROJ * D_MODEL * 2);
    __hip_bfloat16* woutb  = (__hip_bfloat16*)carve((size_t)D_MODEL * D_INNER * 2);
    float*          xIOw   = (float*)carve((size_t)BT * D_IN_PROJ * 4);
    float*          Oo     = (float*)carve((size_t)BT * D_INNER * 4);
    float*          gbuf   = (float*)carve((size_t)BT * N_HEADS * 4);
    float*          whbuf  = (float*)carve((size_t)BT * N_HEADS * 4);
    float*          abuf   = (float*)carve((size_t)BT * N_HEADS * 4);
    __hip_bfloat16* innerb = (__hip_bfloat16*)carve((size_t)BT * D_INNER * 2);

    // 1) convert operands to bf16
    {
        int n4 = (BT * D_MODEL) / 4;
        cvt_bf16_k<<<(n4 + 255) / 256, 256, 0, stream>>>(
            (const float4*)x, (uint2*)xb, n4);
        n4 = (D_IN_PROJ * D_MODEL) / 4;
        cvt_bf16_k<<<(n4 + 255) / 256, 256, 0, stream>>>(
            (const float4*)W_in, (uint2*)winb, n4);
        n4 = (D_MODEL * D_INNER) / 4;
        cvt_bf16_k<<<(n4 + 255) / 256, 256, 0, stream>>>(
            (const float4*)W_out, (uint2*)woutb, n4);
    }

    // 2) xIOw = x @ W_in^T   (M=4096, N=3088, K=1024)
    //    tiles 0..191 with NT=4 (branch-free), ragged tile 192 with NT=1
    {
        dim3 grid_bulk(192 / 4, (BT / 16) / 4);        // (48, 64)
        gemm_bf16_wmma<4><<<grid_bulk, 128, 0, stream>>>(
            xb, winb, xIOw, 0, D_MODEL, D_MODEL, D_MODEL, D_IN_PROJ);
        dim3 grid_tail(1, (BT / 16) / 4);              // (1, 64)
        gemm_bf16_wmma<1><<<grid_tail, 128, 0, stream>>>(
            xb, winb, xIOw, 192, D_MODEL, D_MODEL, D_MODEL, D_IN_PROJ);
    }

    // 3) depthwise conv + gate reduction + wh
    conv_gate_k<<<BT, 1024, 0, stream>>>(xIOw, conv_w, conv_b, w_base,
                                         gbuf, whbuf, Oo);

    // 4) online-softmax scan -> a
    scan_k<<<BB * N_HEADS, 32, 0, stream>>>(whbuf, gbuf, abuf);

    // 5) inner = a * Oo (bf16)
    {
        int n = BT * D_INNER;
        inner_k<<<(n + 255) / 256, 256, 0, stream>>>(abuf, Oo, innerb);
    }

    // 6) out = inner @ W_out^T   (M=4096, N=1024, K=1024)
    {
        dim3 grid((D_MODEL / 16) / 4, (BT / 16) / 4);  // (16, 64)
        gemm_bf16_wmma<4><<<grid, 128, 0, stream>>>(
            innerb, woutb, out, 0, D_INNER, D_INNER, D_INNER, D_MODEL);
    }
    (void)in_sizes; (void)n_in; (void)out_size; (void)ws_size;
}
